// uBlock_15633680957614
// MI455X (gfx1250) — matrix operand order repaired and checked by reading, not compiled, not run on hardware
//
#include <hip/hip_runtime.h>

typedef __attribute__((ext_vector_type(16))) _Float16 v16h;
typedef __attribute__((ext_vector_type(8)))  float    v8f;

#define C_  768
#define T_  2048
#define H_  12
#define D_  64
#define BT_ 4096   // B*T
#define BH_ 24     // B*H
#define LNT 7.6246189861593985f  // ln(2048)

// ---------- optional CDNA5 async global->LDS path (guarded, with fallback) ----------
#if defined(__has_builtin)
# if __has_builtin(__builtin_amdgcn_global_load_async_to_lds_b128)
#  define USE_ASYNC_LDS 1
# endif
#endif
#ifndef USE_ASYNC_LDS
# define USE_ASYNC_LDS 0
#endif

#if USE_ASYNC_LDS
typedef int v4i __attribute__((__vector_size__(4 * sizeof(int))));
typedef __attribute__((address_space(1))) v4i as1_v4i;
typedef __attribute__((address_space(3))) v4i as3_v4i;

__device__ __forceinline__ void async_cp_b128(const _Float16* g, _Float16* l) {
  __builtin_amdgcn_global_load_async_to_lds_b128(
      (as1_v4i*)(uintptr_t)g, (as3_v4i*)(uintptr_t)l, 0, 0);
}
__device__ __forceinline__ void wait_async0() {
# if __has_builtin(__builtin_amdgcn_s_wait_asynccnt)
  __builtin_amdgcn_s_wait_asynccnt(0);
# else
  asm volatile("s_wait_asynccnt 0" ::: "memory");
# endif
}
#endif

// ---------- WMMA fragment loaders (all contiguous -> ds_load_b128 pairs) ----------
// A-layout (16x32 f16): lane half hf, element i<8 -> K=i+hf*8 ; i>=8 -> K=i+8+hf*8
// => two contiguous 8-half runs at colb and colb+16, colb = k0 + hf*8.
template <int STR>
__device__ __forceinline__ v16h load_frag_a(const _Float16 (*src)[STR], int row, int colb) {
  union { v16h v; uint4 u[2]; } t;
  t.u[0] = *(const uint4*)&src[row][colb];
  t.u[1] = *(const uint4*)&src[row][colb + 16];
  return t.v;
}
// B-layout (32x16 f16): lane holds column, same K order as A => runs at colb and colb+16.
template <int STR>
__device__ __forceinline__ v16h load_frag_b(const _Float16 (*src)[STR], int row, int colb) {
  union { v16h v; uint4 u[2]; } t;
  t.u[0] = *(const uint4*)&src[row][colb];
  t.u[1] = *(const uint4*)&src[row][colb + 16];
  return t.v;
}
// Transpose-store: two consecutive k-rows (8 n each) -> dst[n][k] as packed b32 writes.
template <int STR>
__device__ __forceinline__ void store_tr_pair(_Float16 (*dst)[STR], int n0, int krow,
                                              uint4 r0, uint4 r1) {
  const unsigned* a = (const unsigned*)&r0;
  const unsigned* b = (const unsigned*)&r1;
  #pragma unroll
  for (int w = 0; w < 4; ++w) {
    unsigned lo0 = a[w] & 0xffffu, hi0 = a[w] >> 16;
    unsigned lo1 = b[w] & 0xffffu, hi1 = b[w] >> 16;
    *(unsigned*)&dst[n0 + 2 * w][krow]     = (lo1 << 16) | lo0;
    *(unsigned*)&dst[n0 + 2 * w + 1][krow] = (hi1 << 16) | hi0;
  }
}

// ---------------- elementwise f32 -> f16 convert ----------------
__global__ void cvt_f16_kernel(const float* __restrict__ in, _Float16* __restrict__ out, int n) {
  int i = blockIdx.x * 256 + threadIdx.x;
  if (i < n) out[i] = (_Float16)in[i];
}

// ---------------- layernorm (C=768) + cast f16 ----------------
__global__ __launch_bounds__(256) void ln_f16_kernel(const float* __restrict__ x,
                                                     const float* __restrict__ w,
                                                     _Float16* __restrict__ out) {
  __shared__ float red[256];
  int tid = threadIdx.x;
  size_t base = (size_t)blockIdx.x * C_;
  float v0 = x[base + tid], v1 = x[base + tid + 256], v2 = x[base + tid + 512];
  red[tid] = v0 + v1 + v2;
  __syncthreads();
  for (int s = 128; s > 0; s >>= 1) { if (tid < s) red[tid] += red[tid + s]; __syncthreads(); }
  float mu = red[0] * (1.0f / C_);
  __syncthreads();
  float c0 = v0 - mu, c1 = v1 - mu, c2 = v2 - mu;
  red[tid] = c0 * c0 + c1 * c1 + c2 * c2;
  __syncthreads();
  for (int s = 128; s > 0; s >>= 1) { if (tid < s) red[tid] += red[tid + s]; __syncthreads(); }
  float inv = rsqrtf(red[0] * (1.0f / C_) + 1e-5f);
  out[base + tid]       = (_Float16)(c0 * inv * w[tid]);
  out[base + tid + 256] = (_Float16)(c1 * inv * w[tid + 256]);
  out[base + tid + 512] = (_Float16)(c2 * inv * w[tid + 512]);
}

// ---------------- pipelined, double-buffered f16 GEMM with fused epilogues ----------------
#define EPI_QKV   0
#define EPI_RESID 1
#define EPI_GELU  2
#define EPI_ACC   3

__global__ __launch_bounds__(128) void gemm_f16_kernel(
    const _Float16* __restrict__ A, const _Float16* __restrict__ Bm,
    int M, int N, int K, int epi,
    float* __restrict__ outf, const float* __restrict__ resid,
    _Float16* __restrict__ out16,
    _Float16* __restrict__ q16, _Float16* __restrict__ k16, _Float16* __restrict__ v16p,
    const float* __restrict__ qm)
{
  __shared__ __align__(16) _Float16 As[2][64][40];  // row-major [m][k], 80B stride
  __shared__ __align__(16) _Float16 Bt[2][64][40];  // transposed  [n][k], 80B stride
  int tid = threadIdx.x;
  int lane = tid & 31, wave = tid >> 5;
  int hf = lane >> 4, lr = lane & 15;
  int wm = (wave >> 1) * 32, wn = (wave & 1) * 32;
  int m0 = blockIdx.y * 64, n0 = blockIdx.x * 64;

  // A cooperative mapping: chunk0 row = tid>>2, chunk1 row = +32; col = (tid&3)*8
  int arow = tid >> 2, acol = (tid & 3) * 8;
  const _Float16* gA0 = A + (size_t)(m0 + arow) * K + acol;
  const _Float16* gA1 = gA0 + (size_t)32 * K;
  // B cooperative mapping: k-pair p = tid>>3 (rows 2p,2p+1), n-chunk = (tid&7)*8
  int bp = tid >> 3, bnc = (tid & 7) * 8;
  const _Float16* gB0 = Bm + (size_t)(2 * bp) * N + n0 + bnc;
  const _Float16* gB1 = gB0 + N;

  int nk = K >> 5;
  uint4 ra0, ra1, rb0, rb1;

  // ---- prologue: tile 0 into buffer 0 ----
#if USE_ASYNC_LDS
  async_cp_b128(gA0, &As[0][arow][acol]);
  async_cp_b128(gA1, &As[0][arow + 32][acol]);
#else
  ra0 = *(const uint4*)gA0;
  ra1 = *(const uint4*)gA1;
#endif
  rb0 = *(const uint4*)gB0;
  rb1 = *(const uint4*)gB1;
#if !USE_ASYNC_LDS
  *(uint4*)&As[0][arow][acol]      = ra0;
  *(uint4*)&As[0][arow + 32][acol] = ra1;
#endif
  store_tr_pair(Bt[0], bnc, 2 * bp, rb0, rb1);
#if USE_ASYNC_LDS
  wait_async0();
#endif
  __syncthreads();

  v8f acc[2][2] = {};
  for (int kk = 0; kk < nk; ++kk) {
    int cur = kk & 1, nxt = cur ^ 1;
    bool have_next = (kk + 1 < nk);
    if (have_next) {
      gA0 += 32; gA1 += 32;
      gB0 += (size_t)32 * N; gB1 += (size_t)32 * N;
#if USE_ASYNC_LDS
      async_cp_b128(gA0, &As[nxt][arow][acol]);
      async_cp_b128(gA1, &As[nxt][arow + 32][acol]);
#else
      ra0 = *(const uint4*)gA0;
      ra1 = *(const uint4*)gA1;
#endif
      rb0 = *(const uint4*)gB0;
      rb1 = *(const uint4*)gB1;
      if (kk + 2 < nk) {   // global_prefetch_b8 two tiles ahead
        __builtin_prefetch(gA0 + 32, 0, 1);
        __builtin_prefetch(gB0 + (size_t)32 * N, 0, 1);
      }
    }
    // ---- compute on current buffer ----
    v16h af[2], bf[2];
    #pragma unroll
    for (int t = 0; t < 2; ++t) {
      af[t] = load_frag_a(As[cur], wm + t * 16 + lr, hf * 8);
      bf[t] = load_frag_b(Bt[cur], wn + t * 16 + lr, hf * 8);
    }
    #pragma unroll
    for (int ti = 0; ti < 2; ++ti)
      #pragma unroll
      for (int tj = 0; tj < 2; ++tj)
        acc[ti][tj] = __builtin_amdgcn_wmma_f32_16x16x32_f16(
            false, af[ti], false, bf[tj], (short)0, acc[ti][tj], false, false);
    // ---- stage next tile (other buffer), then one barrier ----
    if (have_next) {
#if !USE_ASYNC_LDS
      *(uint4*)&As[nxt][arow][acol]      = ra0;
      *(uint4*)&As[nxt][arow + 32][acol] = ra1;
#endif
      store_tr_pair(Bt[nxt], bnc, 2 * bp, rb0, rb1);
#if USE_ASYNC_LDS
      wait_async0();
#endif
    }
    __syncthreads();
  }

  // ---- epilogue: element (ti,tj,j) -> row m0+wm+ti*16+hf*8+j, col n0+wn+tj*16+lr ----
  #pragma unroll
  for (int ti = 0; ti < 2; ++ti)
  #pragma unroll
  for (int tj = 0; tj < 2; ++tj)
  #pragma unroll
  for (int j = 0; j < 8; ++j) {
    int m = m0 + wm + ti * 16 + hf * 8 + j;
    int n = n0 + wn + tj * 16 + lr;
    float val = acc[ti][tj][j];
    if (epi == EPI_QKV) {
      int which = n / C_;
      int cc = n - which * C_;
      int hd = cc >> 6, d = cc & 63;
      int bb = m >> 11, t = m & 2047;
      size_t di = (((size_t)bb * H_ + hd) * T_ + t) * D_ + d;
      if (which == 0)      q16[di]  = (_Float16)(val * LNT * qm[d]);
      else if (which == 1) k16[di]  = (_Float16)val;
      else                 v16p[di] = (_Float16)val;
    } else if (epi == EPI_RESID) {
      size_t o = (size_t)m * N + n;
      outf[o] = resid[o] + val;
    } else if (epi == EPI_GELU) {
      float g = 0.5f * val * (1.0f + erff(val * 0.70710678118654752f));
      out16[(size_t)m * N + n] = (_Float16)g;
    } else { // EPI_ACC
      size_t o = (size_t)m * N + n;
      outf[o] = outf[o] + val;
    }
  }
}

// ---------------- attention pass 1: per-tile sum of squares of masked scores ----------------
__global__ __launch_bounds__(128) void attn_ss_kernel(
    const _Float16* __restrict__ q, const _Float16* __restrict__ k,
    float* __restrict__ partial)
{
  int qt = blockIdx.x, kt = blockIdx.y, bh = blockIdx.z;
  int pidx = (bh * 32 + kt) * 32 + qt;
  int tid = threadIdx.x;
  if (kt > qt) { if (tid == 0) partial[pidx] = 0.0f; return; }
  __shared__ __align__(16) _Float16 Qs[64][72];
  __shared__ __align__(16) _Float16 Ks[64][72];
  __shared__ float red[128];
  const _Float16* qb = q + (size_t)bh * T_ * D_ + (size_t)qt * 64 * D_;
  const _Float16* kb = k + (size_t)bh * T_ * D_ + (size_t)kt * 64 * D_;
  #pragma unroll
  for (int it = 0; it < 4; ++it) {
    int idx = tid + it * 128;
    int r = idx >> 3, c = (idx & 7) * 8;
#if USE_ASYNC_LDS
    async_cp_b128(&qb[(size_t)r * D_ + c], &Qs[r][c]);
    async_cp_b128(&kb[(size_t)r * D_ + c], &Ks[r][c]);
#else
    *(uint4*)&Qs[r][c] = *(const uint4*)&qb[(size_t)r * D_ + c];
    *(uint4*)&Ks[r][c] = *(const uint4*)&kb[(size_t)r * D_ + c];
#endif
  }
#if USE_ASYNC_LDS
  wait_async0();
#endif
  __syncthreads();
  int lane = tid & 31, wave = tid >> 5, hf = lane >> 4, lr = lane & 15;
  int wm = (wave >> 1) * 32, wn = (wave & 1) * 32;
  v8f acc[2][2] = {};
  #pragma unroll
  for (int k0 = 0; k0 < 64; k0 += 32) {
    v16h af[2], bf[2];
    #pragma unroll
    for (int t = 0; t < 2; ++t) {
      af[t] = load_frag_a(Qs, wm + t * 16 + lr, k0 + hf * 8);
      bf[t] = load_frag_b(Ks, wn + t * 16 + lr, k0 + hf * 8); // B=K^T: contiguous d
    }
    #pragma unroll
    for (int ti = 0; ti < 2; ++ti)
      #pragma unroll
      for (int tj = 0; tj < 2; ++tj)
        acc[ti][tj] = __builtin_amdgcn_wmma_f32_16x16x32_f16(
            false, af[ti], false, bf[tj], (short)0, acc[ti][tj], false, false);
  }
  float local = 0.0f;
  #pragma unroll
  for (int ti = 0; ti < 2; ++ti)
  #pragma unroll
  for (int tj = 0; tj < 2; ++tj)
  #pragma unroll
  for (int j = 0; j < 8; ++j) {
    int qi = qt * 64 + wm + ti * 16 + hf * 8 + j;
    int kj = kt * 64 + wn + tj * 16 + lr;
    float s = (kj <= qi) ? acc[ti][tj][j] * 0.125f : 0.0f;
    local += s * s;
  }
  red[tid] = local;
  __syncthreads();
  for (int s = 64; s > 0; s >>= 1) { if (tid < s) red[tid] += red[tid + s]; __syncthreads(); }
  if (tid == 0) partial[pidx] = red[0];
}

// deterministic fixed-order reduction of the per-tile partials
__global__ __launch_bounds__(256) void reduce_ss_kernel(const float* __restrict__ partial,
                                                        float* __restrict__ sumsq, int n) {
  __shared__ float red[256];
  float s = 0.0f;
  for (int i = threadIdx.x; i < n; i += 256) s += partial[i];
  red[threadIdx.x] = s;
  __syncthreads();
  for (int st = 128; st > 0; st >>= 1) { if (threadIdx.x < st) red[threadIdx.x] += red[threadIdx.x + st]; __syncthreads(); }
  if (threadIdx.x == 0) sumsq[0] = red[0];
}

// ---------------- attention pass 2: recompute scores, scale, fuse att@V ----------------
__global__ __launch_bounds__(128) void attn_av_kernel(
    const _Float16* __restrict__ q, const _Float16* __restrict__ k,
    const _Float16* __restrict__ v, const float* __restrict__ sumsq,
    _Float16* __restrict__ y16)
{
  int qt = blockIdx.x, bh = blockIdx.y;
  int bb = bh / H_, hd = bh - bb * H_;
  __shared__ __align__(16) _Float16 Qs[64][72];
  __shared__ __align__(16) _Float16 Ks[2][64][72];  // row-major [key][d]
  __shared__ __align__(16) _Float16 Vt[2][64][72];  // transposed [d][key]
  __shared__ __align__(16) _Float16 Ss[64][72];
  const _Float16* qb = q + (size_t)bh * T_ * D_ + (size_t)qt * 64 * D_;
  const _Float16* kb = k + (size_t)bh * T_ * D_;
  const _Float16* vb = v + (size_t)bh * T_ * D_;
  int tid = threadIdx.x;
  int lane = tid & 31, wave = tid >> 5, hf = lane >> 4, lr = lane & 15;
  int wm = (wave >> 1) * 32, wn = (wave & 1) * 32;
  // K/Q cooperative mapping (row-major): r = idx>>3, c = (idx&7)*8, 4 chunks
  // V transpose mapping: k-pair vp (+16 for 2nd group), d-chunk vdc
  int vp = tid >> 3, vdc = (tid & 7) * 8;

  // ---- load Q tile + K/V tile kt=0 into buffer 0 ----
  uint4 rv[2][2];
  #pragma unroll
  for (int it = 0; it < 4; ++it) {
    int idx = tid + it * 128;
    int r = idx >> 3, c = (idx & 7) * 8;
#if USE_ASYNC_LDS
    async_cp_b128(&qb[(size_t)r * D_ + c], &Qs[r][c]);
    async_cp_b128(&kb[(size_t)r * D_ + c], &Ks[0][r][c]);
#else
    *(uint4*)&Qs[r][c]    = *(const uint4*)&qb[(size_t)r * D_ + c];
    *(uint4*)&Ks[0][r][c] = *(const uint4*)&kb[(size_t)r * D_ + c];
#endif
  }
  #pragma unroll
  for (int g = 0; g < 2; ++g) {
    int krow = 2 * (vp + g * 16);
    rv[g][0] = *(const uint4*)&vb[(size_t)krow * D_ + vdc];
    rv[g][1] = *(const uint4*)&vb[(size_t)(krow + 1) * D_ + vdc];
  }
  #pragma unroll
  for (int g = 0; g < 2; ++g)
    store_tr_pair(Vt[0], vdc, 2 * (vp + g * 16), rv[g][0], rv[g][1]);
#if USE_ASYNC_LDS
  wait_async0();
#endif
  __syncthreads();

  float scale = 0.125f * rsqrtf(sumsq[0]);
  v8f yacc[2][2] = {};
  for (int kt = 0; kt <= qt; ++kt) {
    int cur = kt & 1, nxt = cur ^ 1;
    bool have_next = (kt + 1 <= qt);
    if (have_next) {
      const _Float16* kbn = kb + (size_t)(kt + 1) * 64 * D_;
      const _Float16* vbn = vb + (size_t)(kt + 1) * 64 * D_;
#if USE_ASYNC_LDS
      #pragma unroll
      for (int it = 0; it < 4; ++it) {
        int idx = tid + it * 128;
        int r = idx >> 3, c = (idx & 7) * 8;
        async_cp_b128(&kbn[(size_t)r * D_ + c], &Ks[nxt][r][c]);
      }
#endif
      #pragma unroll
      for (int g = 0; g < 2; ++g) {
        int krow = 2 * (vp + g * 16);
        rv[g][0] = *(const uint4*)&vbn[(size_t)krow * D_ + vdc];
        rv[g][1] = *(const uint4*)&vbn[(size_t)(krow + 1) * D_ + vdc];
      }
#if !USE_ASYNC_LDS
      #pragma unroll
      for (int it = 0; it < 4; ++it) {
        int idx = tid + it * 128;
        int r = idx >> 3, c = (idx & 7) * 8;
        *(uint4*)&Ks[nxt][r][c] = *(const uint4*)&kbn[(size_t)r * D_ + c];
      }
#endif
    }
    // ---- S = Q @ K^T on current buffer ----
    v8f sacc[2][2] = {};
    #pragma unroll
    for (int k0 = 0; k0 < 64; k0 += 32) {
      v16h af[2], bf[2];
      #pragma unroll
      for (int t = 0; t < 2; ++t) {
        af[t] = load_frag_a(Qs, wm + t * 16 + lr, k0 + hf * 8);
        bf[t] = load_frag_b(Ks[cur], wn + t * 16 + lr, k0 + hf * 8);
      }
      #pragma unroll
      for (int ti = 0; ti < 2; ++ti)
        #pragma unroll
        for (int tj = 0; tj < 2; ++tj)
          sacc[ti][tj] = __builtin_amdgcn_wmma_f32_16x16x32_f16(
              false, af[ti], false, bf[tj], (short)0, sacc[ti][tj], false, false);
    }
    // ---- mask + global-norm scale, stage S as f16 A-matrix ----
    #pragma unroll
    for (int ti = 0; ti < 2; ++ti)
    #pragma unroll
    for (int tj = 0; tj < 2; ++tj)
    #pragma unroll
    for (int j = 0; j < 8; ++j) {
      int lrow = wm + ti * 16 + hf * 8 + j;
      int lcol = wn + tj * 16 + lr;
      int qi = qt * 64 + lrow, kj = kt * 64 + lcol;
      float sv = (kj <= qi) ? sacc[ti][tj][j] * scale : 0.0f;
      Ss[lrow][lcol] = (_Float16)sv;
    }
    __syncthreads();
    // ---- Y += S @ V (Vt is [d][key] so fragments are contiguous) ----
    #pragma unroll
    for (int k0 = 0; k0 < 64; k0 += 32) {
      v16h af[2], bf[2];
      #pragma unroll
      for (int t = 0; t < 2; ++t) {
        af[t] = load_frag_a(Ss, wm + t * 16 + lr, k0 + hf * 8);
        bf[t] = load_frag_b(Vt[cur], wn + t * 16 + lr, k0 + hf * 8);
      }
      #pragma unroll
      for (int ti = 0; ti < 2; ++ti)
        #pragma unroll
        for (int tj = 0; tj < 2; ++tj)
          yacc[ti][tj] = __builtin_amdgcn_wmma_f32_16x16x32_f16(
              false, af[ti], false, bf[tj], (short)0, yacc[ti][tj], false, false);
    }
    // ---- stage next V (transposed) into other buffer ----
    if (have_next) {
      #pragma unroll
      for (int g = 0; g < 2; ++g)
        store_tr_pair(Vt[nxt], vdc, 2 * (vp + g * 16), rv[g][0], rv[g][1]);
#if USE_ASYNC_LDS
      wait_async0();
#endif
    }
    __syncthreads();
  }
  // ---- write y rearranged to [B*T, C] f16 for the cproj GEMM ----
  #pragma unroll
  for (int ti = 0; ti < 2; ++ti)
  #pragma unroll
  for (int tj = 0; tj < 2; ++tj)
  #pragma unroll
  for (int j = 0; j < 8; ++j) {
    int lrow = wm + ti * 16 + hf * 8 + j;
    int lcol = wn + tj * 16 + lr;
    size_t row = (size_t)bb * T_ + qt * 64 + lrow;
    y16[row * C_ + hd * 64 + lcol] = (_Float16)yacc[ti][tj][j];
  }
}

// ---------------- host orchestration ----------------
extern "C" void kernel_launch(void* const* d_in, const int* in_sizes, int n_in,
                              void* d_out, int out_size, void* d_ws, size_t ws_size,
                              hipStream_t stream) {
  (void)in_sizes; (void)n_in; (void)out_size; (void)ws_size;
  const float* x       = (const float*)d_in[0];
  const float* w_attn  = (const float*)d_in[1];
  const float* w_cproj = (const float*)d_in[2];
  const float* qm      = (const float*)d_in[3];
  const float* w_ln1   = (const float*)d_in[4];
  const float* w_ln2   = (const float*)d_in[5];
  const float* w_fc    = (const float*)d_in[6];
  const float* w_mproj = (const float*)d_in[7];
  float* out = (float*)d_out;

  size_t off = 0;
  char* wsb = (char*)d_ws;
  auto alc = [&](size_t bytes) -> void* {
    void* p = wsb + off;
    off = (off + bytes + 255) & ~(size_t)255;
    return p;
  };
  _Float16* h16   = (_Float16*)alc((size_t)BT_ * C_ * 2);
  _Float16* wat16 = (_Float16*)alc((size_t)C_ * 3 * C_ * 2);
  _Float16* wcp16 = (_Float16*)alc((size_t)C_ * C_ * 2);
  _Float16* wfc16 = (_Float16*)alc((size_t)C_ * 4 * C_ * 2);
  _Float16* wmp16 = (_Float16*)alc((size_t)4 * C_ * C_ * 2);
  _Float16* q16   = (_Float16*)alc((size_t)BH_ * T_ * D_ * 2);
  _Float16* k16   = (_Float16*)alc((size_t)BH_ * T_ * D_ * 2);
  _Float16* v16b  = (_Float16*)alc((size_t)BH_ * T_ * D_ * 2);
  _Float16* y16   = (_Float16*)alc((size_t)BT_ * C_ * 2);
  _Float16* h2_16 = (_Float16*)alc((size_t)BT_ * C_ * 2);
  _Float16* a16   = (_Float16*)alc((size_t)BT_ * 4 * C_ * 2);
  float*    part  = (float*)alc((size_t)32 * 32 * BH_ * 4);
  float*    ssq   = (float*)alc(4);

  // weight conversions f32 -> f16
  cvt_f16_kernel<<<(C_ * 3 * C_ + 255) / 256, 256, 0, stream>>>(w_attn, wat16, C_ * 3 * C_);
  cvt_f16_kernel<<<(C_ * C_ + 255) / 256,     256, 0, stream>>>(w_cproj, wcp16, C_ * C_);
  cvt_f16_kernel<<<(C_ * 4 * C_ + 255) / 256, 256, 0, stream>>>(w_fc, wfc16, C_ * 4 * C_);
  cvt_f16_kernel<<<(4 * C_ * C_ + 255) / 256, 256, 0, stream>>>(w_mproj, wmp16, 4 * C_ * C_);

  // ln1
  ln_f16_kernel<<<BT_, 256, 0, stream>>>(x, w_ln1, h16);

  // qkv = ln1(x) @ w_attn, fused split/scale/cast into q16,k16,v16 [B,H,T,D]
  gemm_f16_kernel<<<dim3(36, 64), 128, 0, stream>>>(
      h16, wat16, BT_, 3 * C_, C_, EPI_QKV,
      nullptr, nullptr, nullptr, q16, k16, v16b, qm);

  // global L2 norm of masked scores (deterministic two-stage reduction)
  attn_ss_kernel<<<dim3(32, 32, BH_), 128, 0, stream>>>(q16, k16, part);
  reduce_ss_kernel<<<1, 256, 0, stream>>>(part, ssq, 32 * 32 * BH_);

  // y = (masked scores / norm) @ v, recomputed flash-style
  attn_av_kernel<<<dim3(32, BH_), 128, 0, stream>>>(q16, k16, v16b, ssq, y16);

  // x1 = x + y @ w_cproj  -> d_out (full overwrite)
  gemm_f16_kernel<<<dim3(12, 64), 128, 0, stream>>>(
      y16, wcp16, BT_, C_, C_, EPI_RESID,
      out, x, nullptr, nullptr, nullptr, nullptr, nullptr);

  // ln2 on x1
  ln_f16_kernel<<<BT_, 256, 0, stream>>>(out, w_ln2, h2_16);

  // a = gelu(ln2(x1) @ w_fc) -> f16
  gemm_f16_kernel<<<dim3(48, 64), 128, 0, stream>>>(
      h2_16, wfc16, BT_, 4 * C_, C_, EPI_GELU,
      nullptr, nullptr, a16, nullptr, nullptr, nullptr, nullptr);

  // out = x1 + a @ w_mproj
  gemm_f16_kernel<<<dim3(12, 64), 128, 0, stream>>>(
      a16, wmp16, BT_, C_, 4 * C_, EPI_ACC,
      out, nullptr, nullptr, nullptr, nullptr, nullptr, nullptr);
}
